// MultiHeadSelfAttention_27977416966339
// MI455X (gfx1250) — matrix operand order repaired and checked by reading, not compiled, not run on hardware
//
#include <hip/hip_runtime.h>
#include <hip/hip_bf16.h>
#include <math.h>
#include <stdint.h>

// MI455X / gfx1250: wave32, WMMA 16x16x32 bf16 -> f32 accumulate, TDM tile DMA
// with ping-pong double buffering (TDM for tile i+1 overlaps WMMA on tile i).
// Pipeline: qkv_gemm -> rope_vmix -> gate -> flash attention -> out_gemm.
// Workspace: Qb/Kb/Vb/attn bf16 (4 x 16MB) + gate fp32 (0.5MB) ~= 68MB.

#define Bq   4
#define Sq   2048
#define Dq   1024
#define Hq   16
#define HDq  64
#define HBq  (Hq * Bq)          // 64 head-batch rows
#define MROWS (Bq * Sq)         // 8192

typedef __bf16 bf16_t;
typedef __bf16 v16bf __attribute__((ext_vector_type(16)));
typedef float  v8f   __attribute__((ext_vector_type(8)));
typedef unsigned int u32x4 __attribute__((ext_vector_type(4)));
typedef int          i32x4 __attribute__((ext_vector_type(4)));
typedef int          i32x8 __attribute__((ext_vector_type(8)));

union Frag { unsigned int u[8]; v16bf v; };

// Native converts: v_cvt_pk_bf16_f32 on gfx1250.
__device__ __forceinline__ unsigned short f2bf(float f) {
    union { __bf16 h; unsigned short u; } c;
    c.h = (__bf16)f;
    return c.u;
}
__device__ __forceinline__ unsigned int pack_bf16(float lo, float hi) {
    union { __bf16 h[2]; unsigned int u; } c;
    c.h[0] = (__bf16)lo; c.h[1] = (__bf16)hi;
    return c.u;
}
__device__ __forceinline__ float bf2f(unsigned short b) {
    union { float f; unsigned int u; } c; c.u = ((unsigned int)b) << 16;
    return c.f;
}

// ---------------------------------------------------------------------------
// Tensor Data Mover: 2D tile (height rows x width elems, row stride in elems)
// from global into LDS. D# layout per CDNA5 ISA ch.8. dsz: 0=1B,1=2B,2=4B.
// ---------------------------------------------------------------------------
#if __has_builtin(__builtin_amdgcn_tensor_load_to_lds)
#define HAVE_TDM 1
#else
#define HAVE_TDM 0
#endif
#if __has_include(<hip/amd_detail/amd_gfx1250_TDM.h>)
#define TDM_ARGS6 1
#else
#define TDM_ARGS6 0
#endif

__device__ __forceinline__ void tdm_load_2d(unsigned int lds_off, const void* gptr,
                                            unsigned int dsz, unsigned int width,
                                            unsigned int height, unsigned int stride) {
#if HAVE_TDM
    unsigned long long ga = (unsigned long long)(uintptr_t)gptr;
    u32x4 g0;
    g0[0] = 1u;                                                   // count=1 (valid user D#)
    g0[1] = lds_off;                                              // lds_addr [63:32]
    g0[2] = (unsigned int)(ga & 0xffffffffu);                     // global_addr lo
    g0[3] = (unsigned int)((ga >> 32) & 0x01ffffffu) | (2u << 30);// addr hi | type=2
    i32x8 g1;
    g1[0] = (int)(dsz << 16);                                     // wg_mask=0 | data_size
    g1[1] = (int)(width << 16);                                   // tensor_dim0[15:0] @ [63:48]
    g1[2] = (int)((width >> 16) | (height << 16));                // dim0 hi | tensor_dim1 lo
    g1[3] = (int)((height >> 16) | (width << 16));                // dim1 hi | tile_dim0
    g1[4] = (int)height;                                          // tile_dim1 (tile_dim2=0)
    g1[5] = (int)stride;                                          // tensor_dim0_stride lo
    g1[6] = 0;
    g1[7] = 0;
    i32x4 z4 = {0, 0, 0, 0};
#if TDM_ARGS6
    i32x8 z8 = {0, 0, 0, 0, 0, 0, 0, 0};
    __builtin_amdgcn_tensor_load_to_lds(g0, g1, z4, z4, z8, 0);
#else
    __builtin_amdgcn_tensor_load_to_lds(g0, g1, z4, z4, 0);
#endif
#else
    (void)lds_off; (void)gptr; (void)dsz; (void)width; (void)height; (void)stride;
#endif
}

__device__ __forceinline__ void wait_tensorcnt0() {
#if HAVE_TDM
#if __has_builtin(__builtin_amdgcn_s_wait_tensorcnt)
    __builtin_amdgcn_s_wait_tensorcnt((short)0);
#else
    asm volatile("s_wait_tensorcnt 0x0" ::: "memory");
#endif
#endif
}
__device__ __forceinline__ void wait_tensorcnt1() {   // allow 1 in flight (ping-pong)
#if HAVE_TDM
#if __has_builtin(__builtin_amdgcn_s_wait_tensorcnt)
    __builtin_amdgcn_s_wait_tensorcnt((short)1);
#else
    asm volatile("s_wait_tensorcnt 0x1" ::: "memory");
#endif
#endif
}

#define LDS_OFF(p) ((unsigned int)(uintptr_t)(p))

// ---------------------------------------------------------------------------
// Kernel 1: QKV = x @ Wqkv  (bf16 WMMA, fp32 accum), scatter into (H*B,S,HD)
// Grid: (MROWS/64, 3072/64) = (128, 48), 128 threads (4 waves).
// A tiles (x, fp32 64x32) stream via double-buffered TDM; W tiles staged
// transposed as bf16 by all threads (overlaps the in-flight DMA).
// ---------------------------------------------------------------------------
__global__ __launch_bounds__(128) void qkv_gemm(const float* __restrict__ x,
                                                const float* __restrict__ Wqkv,
                                                unsigned short* __restrict__ qb,
                                                unsigned short* __restrict__ kbuf,
                                                unsigned short* __restrict__ vbuf) {
    __shared__ float lds_af32[2][64 * 32];      // ping-pong TDM destinations
    __shared__ unsigned short lds_wt[64 * 32];  // [col][k] bf16 (W transposed)
    const int t = threadIdx.x;
    const int lane = t & 31, wave = t >> 5;
    const int hh = lane >> 4, nl = lane & 15;
    const int m0 = blockIdx.x * 64;
    const int c0 = blockIdx.y * 64;
    const int NT = Dq / 32;
    v8f acc[4] = {};

#if HAVE_TDM
    if (wave == 0) {   // prime the pipeline: tile 0 -> buffer 0
        tdm_load_2d(LDS_OFF(&lds_af32[0][0]), x + (size_t)m0 * Dq,
                    /*dsz=4B*/2u, 32u, 64u, Dq);
    }
#endif
    for (int it = 0; it < NT; ++it) {
        const int k0 = it * 32;
        const int cur = it & 1;
#if HAVE_TDM
        if (wave == 0) {
            if (it + 1 < NT) {   // issue next tile into the other buffer, then
                tdm_load_2d(LDS_OFF(&lds_af32[cur ^ 1][0]),
                            x + (size_t)m0 * Dq + k0 + 32, 2u, 32u, 64u, Dq);
                wait_tensorcnt1();   // tile `it` done; tile `it+1` still streaming
            } else {
                wait_tensorcnt0();
            }
        }
#else
        {   // fallback: cooperative fp32 copy, 16 elems / thread
            const int row = t >> 1, cb = (t & 1) * 16;
            const float* src = x + (size_t)(m0 + row) * Dq + k0 + cb;
            float* dst = &lds_af32[cur][row * 32 + cb];
#pragma unroll
            for (int i = 0; i < 16; i += 4) *(float4*)(dst + i) = *(const float4*)(src + i);
        }
#endif
        // stage W transposed: W[k][c] -> lds_wt[c][k] (native f32->bf16 cvt)
        {
            const int kk = t >> 2, cb = (t & 3) * 16;
            const float* src = Wqkv + (size_t)(k0 + kk) * 3072 + c0 + cb;
#pragma unroll
            for (int i = 0; i < 16; i += 4) {
                float4 v = *(const float4*)(src + i);
                lds_wt[(cb + i + 0) * 32 + kk] = f2bf(v.x);
                lds_wt[(cb + i + 1) * 32 + kk] = f2bf(v.y);
                lds_wt[(cb + i + 2) * 32 + kk] = f2bf(v.z);
                lds_wt[(cb + i + 3) * 32 + kk] = f2bf(v.w);
            }
        }
        __syncthreads();
        if (k0 + 32 < Dq) { // global_prefetch_b8 for the next W tile
            __builtin_prefetch(Wqkv + (size_t)(k0 + 32 + (t >> 2)) * 3072 + c0, 0, 1);
        }
        // A fragment: 16x32, lane row = nl, half hh; convert fp32 pairs -> bf16
        Frag af;
        {
            const float* arow = &lds_af32[cur][(wave * 16 + nl) * 32];
#pragma unroll
            for (int r = 0; r < 8; ++r) {
                int kk = hh * 16 + r * 2;
                af.u[r] = pack_bf16(arow[kk], arow[kk + 1]);
            }
        }
#pragma unroll
        for (int nb = 0; nb < 4; ++nb) {
            Frag bfr;
            const unsigned int* bcol = (const unsigned int*)&lds_wt[(nb * 16 + nl) * 32];
#pragma unroll
            for (int r = 0; r < 8; ++r) bfr.u[r] = bcol[(hh * 16 + 2 * r) >> 1];
            acc[nb] = __builtin_amdgcn_wmma_f32_16x16x32_bf16(
                false, af.v, false, bfr.v, (short)0, acc[nb], false, false);
        }
        __syncthreads();
    }
    // epilogue: scatter into head-major Q/K/V (bf16)
#pragma unroll
    for (int nb = 0; nb < 4; ++nb) {
#pragma unroll
        for (int r = 0; r < 8; ++r) {
            const int m = m0 + wave * 16 + r + 8 * hh;         // b*S+s
            const int c = c0 + nb * 16 + nl;                   // column in [0,3072)
            const int mat = c >> 10, d = c & 1023;
            const int head = d >> 6, hd = d & 63;
            const int b = m >> 11, s = m & 2047;
            unsigned short* dst = (mat == 0) ? qb : (mat == 1) ? kbuf : vbuf;
            dst[(((size_t)head * Bq + b) * Sq + s) * HDq + hd] = f2bf(acc[nb][r]);
        }
    }
}

// ---------------------------------------------------------------------------
// Kernel 2: in-place RoPE on Q/K + V mixing; writes fp32 V output.
// ---------------------------------------------------------------------------
__global__ __launch_bounds__(128) void rope_vmix(unsigned short* __restrict__ qb,
                                                 unsigned short* __restrict__ kbuf,
                                                 unsigned short* __restrict__ vbuf,
                                                 const float* __restrict__ v1,
                                                 const int* __restrict__ tp,
                                                 const float* __restrict__ alpha1,
                                                 const float* __restrict__ alpha2,
                                                 const float* __restrict__ scale,
                                                 float* __restrict__ vout) {
    const int row = blockIdx.x * 4 + (threadIdx.x >> 5);   // n*S + s, n = h*B + b
    const int lane = threadIdx.x & 31;
    const int s = row & (Sq - 1);
    const int b = (row >> 11) & (Bq - 1);
    const int pos = tp[b * Sq + s];
    const float inv_freq = __expf(-logf(10000.0f) * (float)(2 * lane) / (float)HDq);
    float sn, cs;
    __sincosf((float)pos * inv_freq, &sn, &cs);
    const size_t base = (size_t)row * HDq + 2 * lane;
    { // Q
        float x1 = bf2f(qb[base]), x2 = bf2f(qb[base + 1]);
        qb[base]     = f2bf(x1 * cs - x2 * sn);
        qb[base + 1] = f2bf(x1 * sn + x2 * cs);
    }
    { // K
        float x1 = bf2f(kbuf[base]), x2 = bf2f(kbuf[base + 1]);
        kbuf[base]     = f2bf(x1 * cs - x2 * sn);
        kbuf[base + 1] = f2bf(x1 * sn + x2 * cs);
    }
    { // V = scale*(a1*V + a2*V1)*rsqrt(a1^2+a2^2+eps)
        const float a1 = *alpha1, a2 = *alpha2, sc = *scale;
        const float nrm = sc * rsqrtf(a1 * a1 + a2 * a2 + 1e-8f);
        float w0 = nrm * (a1 * bf2f(vbuf[base])     + a2 * v1[base]);
        float w1 = nrm * (a1 * bf2f(vbuf[base + 1]) + a2 * v1[base + 1]);
        vbuf[base] = f2bf(w0); vbuf[base + 1] = f2bf(w1);
        vout[base] = w0;       vout[base + 1] = w1;
    }
}

// ---------------------------------------------------------------------------
// Kernel 3: gate[b,s,h] = sigmoid(x_row . Wgate[:,h]); one block per row.
// ---------------------------------------------------------------------------
__global__ __launch_bounds__(128) void gate_kernel(const float* __restrict__ x,
                                                   const float* __restrict__ Wg,
                                                   float* __restrict__ gate) {
    __shared__ float red[128];
    const int m = blockIdx.x, t = threadIdx.x;
    const int h = t & 15, seg = t >> 4;
    const float* xr = x + (size_t)m * Dq + seg * 128;
    float acc = 0.f;
    for (int k = 0; k < 128; ++k) acc += xr[k] * Wg[(size_t)(seg * 128 + k) * Hq + h];
    red[t] = acc;
    __syncthreads();
    if (t < Hq) {
        float sum = 0.f;
#pragma unroll
        for (int j = 0; j < 8; ++j) sum += red[t + 16 * j];
        gate[(size_t)m * Hq + t] = 1.0f / (1.0f + __expf(-sum));
    }
}

// ---------------------------------------------------------------------------
// Kernel 4: flash attention (causal, online softmax), gated epilogue.
// Grid: (HB=64, S/64=32), 128 threads; each wave owns a 16-query tile.
// ---------------------------------------------------------------------------
__global__ __launch_bounds__(128) void attn_kernel(const unsigned short* __restrict__ qb,
                                                   const unsigned short* __restrict__ kbuf,
                                                   const unsigned short* __restrict__ vbuf,
                                                   const float* __restrict__ gate,
                                                   unsigned short* __restrict__ ag) {
    __shared__ unsigned short lds_p[4][16 * 32];          // per-wave P tile (bf16)
    const int t = threadIdx.x, lane = t & 31, wave = t >> 5;
    const int hh = lane >> 4, nl = lane & 15;
    const int n = blockIdx.x;                              // head-batch row
    const int q0 = blockIdx.y * 64 + wave * 16;
    const int head = n >> 2, b = n & (Bq - 1);
    const size_t nbase = (size_t)n * Sq * HDq;

    // preload Q A-fragments for both 32-wide K-dim steps of HD=64
    Frag qf[2];
    {
        const unsigned int* qrow = (const unsigned int*)(qb + nbase + (size_t)(q0 + nl) * HDq);
#pragma unroll
        for (int ks = 0; ks < 2; ++ks)
#pragma unroll
            for (int r = 0; r < 8; ++r) {
                int kk = ks * 32 + hh * 16 + r * 2;
                qf[ks].u[r] = qrow[kk >> 1];
            }
    }

    v8f o[4] = {};
    float rmax[8], rsum[8];
#pragma unroll
    for (int r = 0; r < 8; ++r) { rmax[r] = -1e30f; rsum[r] = 0.f; }

    const int kend = q0 + 16;                              // keys <= q0+15
    for (int k0 = 0; k0 < kend; k0 += 32) {
        // ---- scores: Q(16x64) @ K^T(64x32) as two 16x16 WMMA outputs ----
        v8f sc[2];
#pragma unroll
        for (int kb2 = 0; kb2 < 2; ++kb2) {
            v8f a = {};
            const unsigned int* krow =
                (const unsigned int*)(kbuf + nbase + (size_t)(k0 + kb2 * 16 + nl) * HDq);
#pragma unroll
            for (int ks = 0; ks < 2; ++ks) {
                Frag bfr;
#pragma unroll
                for (int r = 0; r < 8; ++r) bfr.u[r] = krow[(ks * 32 + hh * 16 + 2 * r) >> 1];
                a = __builtin_amdgcn_wmma_f32_16x16x32_bf16(
                    false, qf[ks].v, false, bfr.v, (short)0, a, false, false);
            }
            sc[kb2] = a;
        }
        // ---- scale, causal mask, online softmax (rows live in-register) ----
        float fac[8];
#pragma unroll
        for (int r = 0; r < 8; ++r) {
            const int q = q0 + r + 8 * hh;
            float sv0 = sc[0][r] * 0.125f;                 // 1/sqrt(64)
            float sv1 = sc[1][r] * 0.125f;
            if (k0 + nl > q)      sv0 = -1e30f;
            if (k0 + 16 + nl > q) sv1 = -1e30f;
            float mx = fmaxf(sv0, sv1);
#pragma unroll
            for (int off = 1; off < 16; off <<= 1) mx = fmaxf(mx, __shfl_xor(mx, off, 32));
            const float mnew = fmaxf(rmax[r], mx);
            const float f = __expf(rmax[r] - mnew);
            const float p0 = __expf(sv0 - mnew);
            const float p1 = __expf(sv1 - mnew);
            float ps = p0 + p1;
#pragma unroll
            for (int off = 1; off < 16; off <<= 1) ps += __shfl_xor(ps, off, 32);
            rsum[r] = rsum[r] * f + ps;
            rmax[r] = mnew;
            fac[r] = f;
            lds_p[wave][(r + 8 * hh) * 32 + nl]      = f2bf(p0);
            lds_p[wave][(r + 8 * hh) * 32 + 16 + nl] = f2bf(p1);
        }
        // ---- rescale O, then O += P(16x32) @ V(32x64) ----
#pragma unroll
        for (int nb = 0; nb < 4; ++nb)
#pragma unroll
            for (int r = 0; r < 8; ++r) o[nb][r] *= fac[r];
        Frag pf;  // reload P in A-layout (same-wave LDS RAW; compiler inserts dscnt wait)
        {
            const unsigned int* prow = (const unsigned int*)&lds_p[wave][nl * 32];
#pragma unroll
            for (int r = 0; r < 8; ++r) {
                int kk = hh * 16 + r * 2;
                pf.u[r] = prow[kk >> 1];
            }
        }
#pragma unroll
        for (int nb = 0; nb < 4; ++nb) {
            Frag vf;
#pragma unroll
            for (int r = 0; r < 8; ++r) {
                const int kr = k0 + hh * 16 + 2 * r;
                unsigned int lo = vbuf[nbase + (size_t)kr * HDq + nb * 16 + nl];
                unsigned int hi = vbuf[nbase + (size_t)(kr + 1) * HDq + nb * 16 + nl];
                vf.u[r] = lo | (hi << 16);
            }
            o[nb] = __builtin_amdgcn_wmma_f32_16x16x32_bf16(
                false, pf.v, false, vf.v, (short)0, o[nb], false, false);
        }
    }
    // ---- epilogue: 1/l, sigmoid gate, write bf16 in (B,S,H,HD) ----
#pragma unroll
    for (int r = 0; r < 8; ++r) {
        const int q = q0 + r + 8 * hh;
        const float g = gate[((size_t)b * Sq + q) * Hq + head] / rsum[r];
#pragma unroll
        for (int nb = 0; nb < 4; ++nb) {
            ag[(((size_t)b * Sq + q) * Hq + head) * HDq + nb * 16 + nl] = f2bf(o[nb][r] * g);
        }
    }
}

// ---------------------------------------------------------------------------
// Kernel 5: attn_out = attn_gated(bf16) @ Wout (fp32 out). Grid (128,16).
// A tiles (bf16 64x32) stream via double-buffered TDM (data_size=2B).
// ---------------------------------------------------------------------------
__global__ __launch_bounds__(128) void out_gemm(const unsigned short* __restrict__ ag,
                                                const float* __restrict__ Wout,
                                                float* __restrict__ out) {
    __shared__ unsigned short lds_a[2][64 * 32];
    __shared__ unsigned short lds_wt[64 * 32];
    const int t = threadIdx.x, lane = t & 31, wave = t >> 5;
    const int hh = lane >> 4, nl = lane & 15;
    const int m0 = blockIdx.x * 64, c0 = blockIdx.y * 64;
    const int NT = Dq / 32;
    v8f acc[4] = {};

#if HAVE_TDM
    if (wave == 0) {
        tdm_load_2d(LDS_OFF(&lds_a[0][0]), ag + (size_t)m0 * Dq,
                    /*dsz=2B*/1u, 32u, 64u, Dq);
    }
#endif
    for (int it = 0; it < NT; ++it) {
        const int k0 = it * 32;
        const int cur = it & 1;
#if HAVE_TDM
        if (wave == 0) {
            if (it + 1 < NT) {
                tdm_load_2d(LDS_OFF(&lds_a[cur ^ 1][0]),
                            ag + (size_t)m0 * Dq + k0 + 32, 1u, 32u, 64u, Dq);
                wait_tensorcnt1();
            } else {
                wait_tensorcnt0();
            }
        }
#else
        { // fallback: cooperative bf16 copy, 8 dwords / thread
            const unsigned int* src = (const unsigned int*)ag;
            unsigned int* dst = (unsigned int*)&lds_a[cur][0];
#pragma unroll
            for (int i = 0; i < 8; ++i) {
                const int idx = t * 8 + i;
                const int row = idx >> 4, cu = idx & 15;
                dst[idx] = src[(size_t)(m0 + row) * (Dq / 2) + (k0 >> 1) + cu];
            }
        }
#endif
        { // stage Wout transposed, native fp32 -> bf16
            const int kk = t >> 2, cb = (t & 3) * 16;
            const float* src = Wout + (size_t)(k0 + kk) * Dq + c0 + cb;
#pragma unroll
            for (int i = 0; i < 16; i += 4) {
                float4 v = *(const float4*)(src + i);
                lds_wt[(cb + i + 0) * 32 + kk] = f2bf(v.x);
                lds_wt[(cb + i + 1) * 32 + kk] = f2bf(v.y);
                lds_wt[(cb + i + 2) * 32 + kk] = f2bf(v.z);
                lds_wt[(cb + i + 3) * 32 + kk] = f2bf(v.w);
            }
        }
        __syncthreads();
        if (k0 + 32 < Dq) {
            __builtin_prefetch(Wout + (size_t)(k0 + 32 + (t >> 2)) * Dq + c0, 0, 1);
        }
        Frag af;
        {
            const unsigned int* arow = (const unsigned int*)&lds_a[cur][(wave * 16 + nl) * 32];
#pragma unroll
            for (int r = 0; r < 8; ++r) {
                int kk = hh * 16 + r * 2;
                af.u[r] = arow[kk >> 1];
            }
        }
#pragma unroll
        for (int nb = 0; nb < 4; ++nb) {
            Frag bfr;
            const unsigned int* bcol = (const unsigned int*)&lds_wt[(nb * 16 + nl) * 32];
#pragma unroll
            for (int r = 0; r < 8; ++r) bfr.u[r] = bcol[(hh * 16 + 2 * r) >> 1];
            acc[nb] = __builtin_amdgcn_wmma_f32_16x16x32_bf16(
                false, af.v, false, bfr.v, (short)0, acc[nb], false, false);
        }
        __syncthreads();
    }
#pragma unroll
    for (int nb = 0; nb < 4; ++nb)
#pragma unroll
        for (int r = 0; r < 8; ++r)
            out[(size_t)(m0 + wave * 16 + r + 8 * hh) * Dq + c0 + nb * 16 + nl] = acc[nb][r];
}

// ---------------------------------------------------------------------------
extern "C" void kernel_launch(void* const* d_in, const int* in_sizes, int n_in,
                              void* d_out, int out_size, void* d_ws, size_t ws_size,
                              hipStream_t stream) {
    (void)in_sizes; (void)n_in; (void)out_size; (void)ws_size;
    const float* x    = (const float*)d_in[0];
    const float* v1   = (const float*)d_in[1];
    const int*   tp   = (const int*)d_in[2];
    const float* Wqkv = (const float*)d_in[3];
    const float* Wout = (const float*)d_in[4];
    const float* Wg   = (const float*)d_in[5];
    const float* a1   = (const float*)d_in[6];
    const float* a2   = (const float*)d_in[7];
    const float* sc   = (const float*)d_in[8];

    float* attn_out = (float*)d_out;                       // 8,388,608 floats
    float* v_out    = attn_out + (size_t)HBq * Sq * HDq;   // 8,388,608 floats

    const size_t SZ = (size_t)HBq * Sq * HDq;              // 8,388,608 elems
    unsigned short* ws16 = (unsigned short*)d_ws;
    unsigned short* qb = ws16;
    unsigned short* kb = ws16 + SZ;
    unsigned short* vb = ws16 + 2 * SZ;
    unsigned short* ag = ws16 + 3 * SZ;
    float* gate = (float*)(ws16 + 4 * SZ);                 // B*S*H floats

    qkv_gemm  <<<dim3(MROWS / 64, 3072 / 64), 128, 0, stream>>>(x, Wqkv, qb, kb, vb);
    rope_vmix <<<dim3((HBq * Sq) / 4),        128, 0, stream>>>(qb, kb, vb, v1, tp, a1, a2, sc, v_out);
    gate_kernel<<<dim3(MROWS),                128, 0, stream>>>(x, Wg, gate);
    attn_kernel<<<dim3(HBq, Sq / 64),         128, 0, stream>>>(qb, kb, vb, gate, ag);
    out_gemm  <<<dim3(MROWS / 64, Dq / 64),   128, 0, stream>>>(ag, Wout, attn_out);
}
